// DateMemoryUnit_20289425506971
// MI455X (gfx1250) — matrix operand order repaired and checked by reading, not compiled, not run on hardware
//
#include <hip/hip_runtime.h>

// Problem constants (fixed by the reference).
#define NN   65536
#define MDIM 2048
#define FDIM 128
#define SHRINK 0.0025f
#define EPSV 1e-12f

typedef __attribute__((ext_vector_type(16))) _Float16 v16h;
typedef __attribute__((ext_vector_type(8)))  float    v8f;

// ---------------------------------------------------------------------------
// Pre-pass: convert mem_unit (f32 [2048,128]) to f16 row-major and f16
// transposed [128,2048] in workspace, so both GEMMs get contiguous B loads.
// ---------------------------------------------------------------------------
__global__ __launch_bounds__(256) void cvt_mem_kernel(const float* __restrict__ mem,
                                                      _Float16* __restrict__ mem16,
                                                      _Float16* __restrict__ mem16t) {
    int i = blockIdx.x * 256 + threadIdx.x;      // 0 .. 2048*128-1
    int m = i >> 7;
    int f = i & 127;
    float v = mem[i];
    mem16[i] = (_Float16)v;
    mem16t[f * MDIM + m] = (_Float16)v;
}

// ---------------------------------------------------------------------------
// Fused kernel: one block = 16 rows of N. 256 threads = 8 wave32 waves.
// ---------------------------------------------------------------------------
#define LDS_STRIDE (MDIM + 16)   // pad 16 floats -> adjacent rows hit disjoint banks

__global__ __launch_bounds__(256) void fused_kernel(
    const float* __restrict__ input,   // [N,128]
    const float* __restrict__ datev,   // [N,2]
    const float* __restrict__ dmap,    // [2048,2]
    const _Float16* __restrict__ mem16,   // [2048,128] f16
    const _Float16* __restrict__ mem16t,  // [128,2048] f16
    float* __restrict__ out,           // [N,128]
    float* __restrict__ att_out) {     // [N,2048]

    __shared__ float sc[16 * LDS_STRIDE];  // fused logits -> e -> h -> att
    __shared__ float red[256];             // per-row reductions (16 rows x 16 slots)

    const int tid  = threadIdx.x;
    const int wave = tid >> 5;
    const int lane = tid & 31;
    const int half = lane >> 4;    // 0: lanes 0-15, 1: lanes 16-31
    const int l16  = lane & 15;
    const int n0   = blockIdx.x * 16;

    // ---- A fragments: input rows n0..n0+15 as f16 (16x32 per chunk, K=128) ----
    // A layout: lanes 0-15 (M=lane): K=[kb..kb+15];
    //           lanes 16-31 (M=lane-16): K=[kb+16..kb+31].
    v16h A[4];
    {
        const float* arow = input + (size_t)(n0 + l16) * FDIM;
        #pragma unroll
        for (int c = 0; c < 4; ++c) {
            int kb = c * 32 + half * 16;
            #pragma unroll
            for (int j = 0; j < 8; ++j) {
                A[c][j]     = (_Float16)arow[kb + j];
                A[c][8 + j] = (_Float16)arow[kb + 8 + j];
            }
        }
    }

    // Date vectors for the 8 M-rows this lane's accumulators map to.
    float dvx[8], dvy[8];
    {
        int mbase = n0 + half * 8;
        #pragma unroll
        for (int r = 0; r < 8; ++r) {
            dvx[r] = datev[(size_t)(mbase + r) * 2 + 0];
            dvy[r] = datev[(size_t)(mbase + r) * 2 + 1];
        }
    }

    // ---- GEMM1: value_att tile + date gating -> logits into LDS ----
    // Each wave covers 256 columns of mem_dim (16 tiles of 16).
    for (int mt = 0; mt < 16; ++mt) {
        const int m0 = wave * 256 + mt * 16;
        // B fragments (32x16 f16): lanes 0-15 (N=lane): K=[kb..kb+15];
        //                          lanes 16-31 (N=lane-16): K=[kb+16..kb+31].
        v16h B[4];
        const _Float16* brow = mem16 + (size_t)(m0 + l16) * FDIM;
        #pragma unroll
        for (int c = 0; c < 4; ++c) {
            int kb = c * 32 + half * 16;
            #pragma unroll
            for (int j = 0; j < 16; ++j) B[c][j] = brow[kb + j];
        }
        v8f acc = {};
        #pragma unroll
        for (int c = 0; c < 4; ++c)
            acc = __builtin_amdgcn_wmma_f32_16x16x32_f16(
                false, A[c], false, B[c], (short)0, acc, false, false);

        // D layout: element r of lane -> (M = half*8 + r, N = l16).
        float dmx = dmap[(size_t)(m0 + l16) * 2 + 0];
        float dmy = dmap[(size_t)(m0 + l16) * 2 + 1];
        #pragma unroll
        for (int r = 0; r < 8; ++r) {
            float da = fmaxf(dvx[r] * dmx + dvy[r] * dmy, 0.0f);
            sc[(half * 8 + r) * LDS_STRIDE + m0 + l16] = acc[r] * da;
        }
    }
    __syncthreads();

    // ---- softmax + hard_shrink_relu + L1 normalize (16 threads per row) ----
    const int row = tid >> 4;
    const int s   = tid & 15;
    float* srow = sc + row * LDS_STRIDE;

    float mx = -3.402823e38f;
    for (int j = s; j < MDIM; j += 16) mx = fmaxf(mx, srow[j]);
    red[tid] = mx;
    __syncthreads();
    mx = red[row * 16];
    #pragma unroll
    for (int k = 1; k < 16; ++k) mx = fmaxf(mx, red[row * 16 + k]);
    __syncthreads();

    float sum = 0.0f;
    for (int j = s; j < MDIM; j += 16) {
        float e = __expf(srow[j] - mx);
        srow[j] = e;
        sum += e;
    }
    red[tid] = sum;
    __syncthreads();
    sum = 0.0f;
    #pragma unroll
    for (int k = 0; k < 16; ++k) sum += red[row * 16 + k];
    __syncthreads();

    const float inv = 1.0f / sum;
    float l1 = 0.0f;
    for (int j = s; j < MDIM; j += 16) {
        float x = srow[j] * inv;                 // softmax value
        float d = x - SHRINK;
        float h = (d > 0.0f) ? (x * d / (d + EPSV)) : 0.0f;  // relu(d)*x/(|d|+eps)
        srow[j] = h;
        l1 += h;
    }
    red[tid] = l1;
    __syncthreads();
    l1 = 0.0f;
    #pragma unroll
    for (int k = 0; k < 16; ++k) l1 += red[row * 16 + k];

    const float invn = 1.0f / fmaxf(l1, EPSV);
    float* attrow = att_out + (size_t)(n0 + row) * MDIM;
    for (int j = s; j < MDIM; j += 16) {
        float a = srow[j] * invn;
        srow[j] = a;        // keep for GEMM2
        attrow[j] = a;      // final att output
    }
    __syncthreads();

    // ---- GEMM2: out[16,128] = att[16,2048] @ mem_unit[2048,128] ----
    // Each wave owns one 16-wide f tile; K = 2048 in 64 chunks of 32.
    const int f0 = wave * 16;
    v8f oc = {};
    for (int kc = 0; kc < 64; ++kc) {
        const int kb = kc * 32;
        v16h a;
        {
            const float* ar = sc + l16 * LDS_STRIDE + kb + half * 16;
            #pragma unroll
            for (int j = 0; j < 8; ++j) {
                a[j]     = (_Float16)ar[j];
                a[8 + j] = (_Float16)ar[8 + j];
            }
        }
        v16h b;
        {
            const _Float16* br = mem16t + (size_t)(f0 + l16) * MDIM + kb + half * 16;
            #pragma unroll
            for (int j = 0; j < 16; ++j) b[j] = br[j];
        }
        oc = __builtin_amdgcn_wmma_f32_16x16x32_f16(
            false, a, false, b, (short)0, oc, false, false);
    }
    // D: (M = half*8 + r, N = l16) -> out[n0+M][f0+l16]
    float* obase = out + (size_t)n0 * FDIM + f0 + l16;
    #pragma unroll
    for (int r = 0; r < 8; ++r) {
        obase[(size_t)(half * 8 + r) * FDIM] = oc[r];
    }
}

// ---------------------------------------------------------------------------
extern "C" void kernel_launch(void* const* d_in, const int* in_sizes, int n_in,
                              void* d_out, int out_size, void* d_ws, size_t ws_size,
                              hipStream_t stream) {
    const float* input = (const float*)d_in[0];  // [65536,128]
    const float* datev = (const float*)d_in[1];  // [65536,2]
    const float* mem   = (const float*)d_in[2];  // [2048,128]
    const float* dmap  = (const float*)d_in[3];  // [2048,2]

    float* out = (float*)d_out;                          // [65536,128]
    float* att = out + (size_t)NN * FDIM;                // [65536,2048]

    _Float16* mem16  = (_Float16*)d_ws;                  // 512 KB
    _Float16* mem16t = mem16 + (size_t)MDIM * FDIM;      // 512 KB

    cvt_mem_kernel<<<(MDIM * FDIM) / 256, 256, 0, stream>>>(mem, mem16, mem16t);
    fused_kernel<<<NN / 16, 256, 0, stream>>>(input, datev, dmap,
                                              mem16, mem16t, out, att);
}
